// SelfAttention1D_19009525252192
// MI455X (gfx1250) — compile-verified, matrix operand order checked by reading, not run on hardware
//
#include <hip/hip_runtime.h>
#include <hip/hip_bf16.h>
#include <stdint.h>

// ---------------------------------------------------------------------------
// MI455X (gfx1250) self-attention, bf16 WMMA pipeline with async-to-LDS
// double-buffered GEMM staging.
//   B=4, N=2048, D=1024.  All GEMMs via v_wmma_f32_16x16x32_bf16.
//   Data staging via global_load_async_to_lds_b128 + s_wait_asynccnt.
// Workspace (assumed ws_size >= ~174 MB):
//   xb[16MB] wq/wk/wvb[6MB] Qb[16MB] Kb[16MB] Vt[16MB] S[64MB] P[32MB]
// ---------------------------------------------------------------------------

typedef __bf16 bf16_t;
typedef __attribute__((ext_vector_type(16))) __bf16 v16bf;
typedef __attribute__((ext_vector_type(8)))  __bf16 v8bf;
typedef __attribute__((ext_vector_type(8)))  float  v8f;

#define BATCH 4
#define SEQ   2048
#define DIM   1024
#define TOKENS (BATCH * SEQ)   // 8192

// Elements per staged tile: 128 rows x 32 k = 4096 bf16 = 8 KB.
#define TILE_ELEMS 4096

// fp32 -> bf16, round-to-nearest-even (pure integer ops)
__device__ __forceinline__ bf16_t f2bf(float f) {
  union { float f; unsigned u; } v; v.f = f;
  unsigned r = v.u + 0x7FFFu + ((v.u >> 16) & 1u);
  unsigned short h = (unsigned short)(r >> 16);
  return __builtin_bit_cast(bf16_t, h);
}

// ----- CDNA5 async global->LDS copy (16 bytes / lane), ASYNCcnt tracked -----
__device__ __forceinline__ void async_copy16(unsigned lds_addr, const void* gaddr) {
  asm volatile("global_load_async_to_lds_b128 %0, %1, off"
               :: "v"(lds_addr), "v"(gaddr)
               : "memory");
}

template <int N>
__device__ __forceinline__ void wait_asynccnt() {
  asm volatile("s_wait_asynccnt %0" :: "n"(N) : "memory");
}

__device__ __forceinline__ unsigned lds_off(const void* p) {
  // Low 32 bits of a generic LDS address == LDS byte offset (ISA aperture map).
  return (unsigned)(uintptr_t)p;
}

// Stage one 128x32 A-tile and one 128x32 B-tile into LDS buffers [buf].
// 256 threads: thread t covers row t>>1, element half (t&1)*16 (32 bytes),
// issuing 2 x b128 async per matrix => 4 async ops per thread per stage.
__device__ __forceinline__ void stage_tiles(const bf16_t* __restrict__ Ag, int lda,
                                            const bf16_t* __restrict__ Bg, int ldb,
                                            bf16_t* As, bf16_t* Bs,
                                            int buf, int k) {
  int t = threadIdx.x;
  int r = t >> 1;
  int h = (t & 1) << 4;                 // 0 or 16 elements
  const bf16_t* ga = Ag + (size_t)r * lda + k + h;
  unsigned la = lds_off(&As[buf * TILE_ELEMS + r * 32 + h]);
  async_copy16(la,       ga);
  async_copy16(la + 16u, ga + 8);
  const bf16_t* gb = Bg + (size_t)r * ldb + k + h;
  unsigned lb = lds_off(&Bs[buf * TILE_ELEMS + r * 32 + h]);
  async_copy16(lb,       gb);
  async_copy16(lb + 16u, gb + 8);
}

// Load one 16x32 bf16 WMMA A/B fragment from an LDS tile [128][32].
// ISA layout: lane L(0..15) row r0+L, K=0..7 & 16..23; lane L(16..31) same
// rows, K=8..15 & 24..31.
__device__ __forceinline__ v16bf load_frag_lds(const bf16_t* tile, int r0) {
  int lane = threadIdx.x & 31;
  int r  = r0 + (lane & 15);
  int ko = (lane >> 4) << 3;            // 0 or 8
  const bf16_t* p = tile + r * 32 + ko;
  v8bf lo = *reinterpret_cast<const v8bf*>(p);        // K = ko .. ko+7
  v8bf hi = *reinterpret_cast<const v8bf*>(p + 16);   // K = ko+16 .. ko+23
  v16bf f;
#pragma unroll
  for (int i = 0; i < 8; ++i) { f[i] = lo[i]; f[i + 8] = hi[i]; }
  return f;
}

__device__ __forceinline__ v8f wmma_bf16(v16bf a, v16bf b, v8f c) {
  return __builtin_amdgcn_wmma_f32_16x16x32_bf16(
      false, a, false, b, (short)0, c, false, false);
}

// Double-buffered 128x128 block GEMM core. Ag/Bg point at the block-tile
// origin rows (A: block rows, B: rows of B^T i.e. [N x K] row-major).
// Wave tile: 32(M) x 64(N) => acc[2][4] of 16x16 f32 tiles.
__device__ __forceinline__ void gemm_core(const bf16_t* __restrict__ Ag, int lda,
                                          const bf16_t* __restrict__ Bg, int ldb,
                                          int Kd, bf16_t* As, bf16_t* Bs,
                                          int wm, int wn, v8f acc[2][4]) {
#pragma unroll
  for (int m = 0; m < 2; ++m)
#pragma unroll
    for (int n = 0; n < 4; ++n) acc[m][n] = (v8f){};

  const int nk = Kd / 32;
  stage_tiles(Ag, lda, Bg, ldb, As, Bs, 0, 0);

  for (int kt = 0; kt < nk; ++kt) {
    int buf = kt & 1;
    if (kt + 1 < nk) {
      stage_tiles(Ag, lda, Bg, ldb, As, Bs, 1 - buf, (kt + 1) * 32);
      wait_asynccnt<4>();      // previous stage (4 per-wave async ops) done
    } else {
      wait_asynccnt<0>();
    }
    __syncthreads();           // staged tiles visible to all waves

    const bf16_t* At = As + buf * TILE_ELEMS;
    const bf16_t* Bt = Bs + buf * TILE_ELEMS;
    v16bf a0 = load_frag_lds(At, wm * 32);
    v16bf a1 = load_frag_lds(At, wm * 32 + 16);
    v16bf b0 = load_frag_lds(Bt, wn * 64);
    v16bf b1 = load_frag_lds(Bt, wn * 64 + 16);
    v16bf b2 = load_frag_lds(Bt, wn * 64 + 32);
    v16bf b3 = load_frag_lds(Bt, wn * 64 + 48);
    acc[0][0] = wmma_bf16(a0, b0, acc[0][0]);
    acc[0][1] = wmma_bf16(a0, b1, acc[0][1]);
    acc[0][2] = wmma_bf16(a0, b2, acc[0][2]);
    acc[0][3] = wmma_bf16(a0, b3, acc[0][3]);
    acc[1][0] = wmma_bf16(a1, b0, acc[1][0]);
    acc[1][1] = wmma_bf16(a1, b1, acc[1][1]);
    acc[1][2] = wmma_bf16(a1, b2, acc[1][2]);
    acc[1][3] = wmma_bf16(a1, b3, acc[1][3]);

    __syncthreads();           // all reads done before buffer re-stage
  }
}

// ---------------------------------------------------------------------------
// fp32 -> bf16 conversion (8 elements per thread)
// ---------------------------------------------------------------------------
__global__ __launch_bounds__(256) void cvt_f32_bf16(const float* __restrict__ in,
                                                    bf16_t* __restrict__ out,
                                                    int n8) {
  int i = blockIdx.x * blockDim.x + threadIdx.x;
  if (i >= n8) return;
  const float4* p = reinterpret_cast<const float4*>(in) + (size_t)i * 2;
  float4 a = p[0];
  float4 b = p[1];
  v8bf o;
  o[0] = f2bf(a.x); o[1] = f2bf(a.y); o[2] = f2bf(a.z); o[3] = f2bf(a.w);
  o[4] = f2bf(b.x); o[5] = f2bf(b.y); o[6] = f2bf(b.z); o[7] = f2bf(b.w);
  reinterpret_cast<v8bf*>(out)[i] = o;
}

// ---------------------------------------------------------------------------
// QKV projection:  C[r,c] = (sum_k A[r,k]*W[c,k] + bias[c]) * scale
// vtrans==0: bf16 to Cn [8192 x 1024]; vtrans==1: V^T store Ct[b*D*N + c*N + n]
// ---------------------------------------------------------------------------
__global__ __launch_bounds__(256) void gemm_qkv(const bf16_t* __restrict__ A,
                                                const bf16_t* __restrict__ W,
                                                const float* __restrict__ bias,
                                                bf16_t* __restrict__ Cn,
                                                bf16_t* __restrict__ Ct,
                                                float scale, int vtrans) {
  __shared__ bf16_t As[2 * TILE_ELEMS];
  __shared__ bf16_t Bs[2 * TILE_ELEMS];
  int lane = threadIdx.x & 31;
  int w    = threadIdx.x >> 5;
  int wm   = w & 3, wn = w >> 2;
  int row0 = blockIdx.y * 128 + wm * 32;
  int col0 = blockIdx.x * 128 + wn * 64;

  v8f acc[2][4];
  gemm_core(A + (size_t)(blockIdx.y * 128) * DIM, DIM,
            W + (size_t)(blockIdx.x * 128) * DIM, DIM,
            DIM, As, Bs, wm, wn, acc);

  int lr = (lane >> 4) << 3;
  int lc = lane & 15;
#pragma unroll
  for (int mt = 0; mt < 2; ++mt)
#pragma unroll
    for (int nt = 0; nt < 4; ++nt)
#pragma unroll
      for (int v = 0; v < 8; ++v) {
        int row = row0 + mt * 16 + lr + v;
        int col = col0 + nt * 16 + lc;
        float val = (acc[mt][nt][v] + bias[col]) * scale;
        bf16_t h = f2bf(val);
        if (!vtrans) {
          Cn[(size_t)row * DIM + col] = h;
        } else {
          int b = row >> 11;
          int n = row & (SEQ - 1);
          Ct[((size_t)b * DIM + col) * SEQ + n] = h;
        }
      }
}

// ---------------------------------------------------------------------------
// scores:  S[b,i,j] = sum_d Q[b,i,d] * K[b,j,d]   (delta pre-folded into Q)
// ---------------------------------------------------------------------------
__global__ __launch_bounds__(256) void gemm_scores(const bf16_t* __restrict__ Q,
                                                   const bf16_t* __restrict__ Kmat,
                                                   float* __restrict__ S) {
  __shared__ bf16_t As[2 * TILE_ELEMS];
  __shared__ bf16_t Bs[2 * TILE_ELEMS];
  int b = blockIdx.z;
  const bf16_t* Ab = Q    + (size_t)b * SEQ * DIM;
  const bf16_t* Bb = Kmat + (size_t)b * SEQ * DIM;
  float*        Sb = S    + (size_t)b * SEQ * SEQ;

  int lane = threadIdx.x & 31;
  int w    = threadIdx.x >> 5;
  int wm   = w & 3, wn = w >> 2;
  int row0 = blockIdx.y * 128 + wm * 32;
  int col0 = blockIdx.x * 128 + wn * 64;

  v8f acc[2][4];
  gemm_core(Ab + (size_t)(blockIdx.y * 128) * DIM, DIM,
            Bb + (size_t)(blockIdx.x * 128) * DIM, DIM,
            DIM, As, Bs, wm, wn, acc);

  int lr = (lane >> 4) << 3;
  int lc = lane & 15;
#pragma unroll
  for (int mt = 0; mt < 2; ++mt)
#pragma unroll
    for (int nt = 0; nt < 4; ++nt)
#pragma unroll
      for (int v = 0; v < 8; ++v) {
        int row = row0 + mt * 16 + lr + v;
        int col = col0 + nt * 16 + lc;
        Sb[(size_t)row * SEQ + col] = acc[mt][nt][v];
      }
}

// ---------------------------------------------------------------------------
// softmax over rows of S (length 2048), bf16 P out. One block (8 waves) / row.
// ---------------------------------------------------------------------------
__global__ __launch_bounds__(256) void softmax_rows(const float* __restrict__ S,
                                                    bf16_t* __restrict__ P) {
  __shared__ float red_m[8];
  __shared__ float red_s[8];
  size_t row = blockIdx.x;
  const float* s = S + row * SEQ;
  bf16_t*      p = P + row * SEQ;
  int t = threadIdx.x;
  int lane = t & 31, wid = t >> 5;

  float4 a = reinterpret_cast<const float4*>(s)[t];
  float4 b = reinterpret_cast<const float4*>(s)[t + 256];

  float m = fmaxf(fmaxf(fmaxf(a.x, a.y), fmaxf(a.z, a.w)),
                  fmaxf(fmaxf(b.x, b.y), fmaxf(b.z, b.w)));
#pragma unroll
  for (int off = 16; off > 0; off >>= 1)
    m = fmaxf(m, __shfl_xor(m, off, 32));
  if (lane == 0) red_m[wid] = m;
  __syncthreads();
  m = red_m[0];
#pragma unroll
  for (int i = 1; i < 8; ++i) m = fmaxf(m, red_m[i]);

  float e[8];
  e[0] = __expf(a.x - m); e[1] = __expf(a.y - m);
  e[2] = __expf(a.z - m); e[3] = __expf(a.w - m);
  e[4] = __expf(b.x - m); e[5] = __expf(b.y - m);
  e[6] = __expf(b.z - m); e[7] = __expf(b.w - m);
  float l = e[0] + e[1] + e[2] + e[3] + e[4] + e[5] + e[6] + e[7];
#pragma unroll
  for (int off = 16; off > 0; off >>= 1)
    l += __shfl_xor(l, off, 32);
  if (lane == 0) red_s[wid] = l;
  __syncthreads();
  l = red_s[0] + red_s[1] + red_s[2] + red_s[3] +
      red_s[4] + red_s[5] + red_s[6] + red_s[7];
  float r = __frcp_rn(l);

#pragma unroll
  for (int i = 0; i < 4; ++i) p[4 * t + i]         = f2bf(e[i] * r);
#pragma unroll
  for (int i = 0; i < 4; ++i) p[4 * (t + 256) + i] = f2bf(e[4 + i] * r);
}

// ---------------------------------------------------------------------------
// out[b,i,e] = gamma * sum_j P[b,i,j] * Vt[b,e,j]  + x[b,i,e]    (fp32 out)
// ---------------------------------------------------------------------------
__global__ __launch_bounds__(256) void gemm_out(const bf16_t* __restrict__ P,
                                                const bf16_t* __restrict__ Vt,
                                                const float* __restrict__ x,
                                                const float* __restrict__ gamma,
                                                float* __restrict__ out) {
  __shared__ bf16_t As[2 * TILE_ELEMS];
  __shared__ bf16_t Bs[2 * TILE_ELEMS];
  int b = blockIdx.z;
  const bf16_t* Ab = P  + (size_t)b * SEQ * SEQ;
  const bf16_t* Bb = Vt + (size_t)b * DIM * SEQ;

  int lane = threadIdx.x & 31;
  int w    = threadIdx.x >> 5;
  int wm   = w & 3, wn = w >> 2;
  int row0 = blockIdx.y * 128 + wm * 32;
  int col0 = blockIdx.x * 128 + wn * 64;

  v8f acc[2][4];
  gemm_core(Ab + (size_t)(blockIdx.y * 128) * SEQ, SEQ,
            Bb + (size_t)(blockIdx.x * 128) * SEQ, SEQ,
            SEQ, As, Bs, wm, wn, acc);

  float g = gamma[0];
  int lr = (lane >> 4) << 3;
  int lc = lane & 15;
#pragma unroll
  for (int mt = 0; mt < 2; ++mt)
#pragma unroll
    for (int nt = 0; nt < 4; ++nt)
#pragma unroll
      for (int v = 0; v < 8; ++v) {
        int row = row0 + mt * 16 + lr + v;
        int col = col0 + nt * 16 + lc;
        size_t gi = ((size_t)b * SEQ + row) * DIM + col;
        out[gi] = acc[mt][nt][v] * g + x[gi];
      }
}

// ---------------------------------------------------------------------------
// host launcher
// ---------------------------------------------------------------------------
extern "C" void kernel_launch(void* const* d_in, const int* in_sizes, int n_in,
                              void* d_out, int out_size, void* d_ws, size_t ws_size,
                              hipStream_t stream) {
  const float* x     = (const float*)d_in[0];
  const float* Wq    = (const float*)d_in[1];
  const float* bq    = (const float*)d_in[2];
  const float* Wk    = (const float*)d_in[3];
  const float* bk    = (const float*)d_in[4];
  const float* Wv    = (const float*)d_in[5];
  const float* bv    = (const float*)d_in[6];
  const float* gamma = (const float*)d_in[7];
  float* out = (float*)d_out;

  char* w = (char*)d_ws;
  bf16_t* xb  = (bf16_t*)w; w += (size_t)TOKENS * DIM * 2;
  bf16_t* wqb = (bf16_t*)w; w += (size_t)DIM * DIM * 2;
  bf16_t* wkb = (bf16_t*)w; w += (size_t)DIM * DIM * 2;
  bf16_t* wvb = (bf16_t*)w; w += (size_t)DIM * DIM * 2;
  bf16_t* Qb  = (bf16_t*)w; w += (size_t)TOKENS * DIM * 2;
  bf16_t* Kb  = (bf16_t*)w; w += (size_t)TOKENS * DIM * 2;
  bf16_t* Vt  = (bf16_t*)w; w += (size_t)TOKENS * DIM * 2;
  float*  S   = (float*)w;  w += (size_t)BATCH * SEQ * SEQ * 4;
  bf16_t* P   = (bf16_t*)w; w += (size_t)BATCH * SEQ * SEQ * 2;

  const float delta = 0.03125f;   // 1/sqrt(1024)

  // 1) fp32 -> bf16 conversions
  {
    int n8 = TOKENS * DIM / 8;
    cvt_f32_bf16<<<(n8 + 255) / 256, 256, 0, stream>>>(x, xb, n8);
    int m8 = DIM * DIM / 8;
    cvt_f32_bf16<<<(m8 + 255) / 256, 256, 0, stream>>>(Wq, wqb, m8);
    cvt_f32_bf16<<<(m8 + 255) / 256, 256, 0, stream>>>(Wk, wkb, m8);
    cvt_f32_bf16<<<(m8 + 255) / 256, 256, 0, stream>>>(Wv, wvb, m8);
  }

  // 2) Q, K, V projections (Q pre-scaled by delta; V stored transposed)
  {
    dim3 grid(DIM / 128, TOKENS / 128, 1);
    gemm_qkv<<<grid, 256, 0, stream>>>(xb, wqb, bq, Qb, nullptr, delta, 0);
    gemm_qkv<<<grid, 256, 0, stream>>>(xb, wkb, bk, Kb, nullptr, 1.0f, 0);
    gemm_qkv<<<grid, 256, 0, stream>>>(xb, wvb, bv, nullptr, Vt, 1.0f, 1);
  }

  // 3) scores S = Q K^T (batched)
  {
    dim3 grid(SEQ / 128, SEQ / 128, BATCH);
    gemm_scores<<<grid, 256, 0, stream>>>(Qb, Kb, S);
  }

  // 4) row softmax -> bf16 P
  softmax_rows<<<BATCH * SEQ, 256, 0, stream>>>(S, P);

  // 5) out = gamma * (P V) + x
  {
    dim3 grid(DIM / 128, SEQ / 128, BATCH);
    gemm_out<<<grid, 256, 0, stream>>>(P, Vt, x, gamma, out);
  }
}
